// TopKGatev2_21646635172054
// MI455X (gfx1250) — matrix operand order repaired and checked by reading, not compiled, not run on hardware
//
#include <hip/hip_runtime.h>
#include <hip/hip_bf16.h>
#include <math.h>

typedef __attribute__((ext_vector_type(16))) __bf16 v16bf;
typedef __attribute__((ext_vector_type(8)))  float  v8f;

#define T_TOK   16384
#define DMODEL  4096
#define NEXP    128
#define TOPK    2
#define CAP     256                 // ceil(16384/128 * 1.0 * 2)
#define NASSIGN (T_TOK*TOPK)        // 32768
#define TOK_PER_BLK 64
#define NBLK1   (T_TOK/TOK_PER_BLK) // 256
#define MAXPER  4096                // per-expert LDS sort capacity

// d_out float offsets (reference tuple return order)
#define O_LAUX 0
#define O_IDX  1
#define O_BIDS (O_IDX  + NASSIGN)   // 32769
#define O_BINS (O_BIDS + NASSIGN)   // 65537
#define O_W    (O_BINS + NEXP)      // 65665
#define O_TPE  (O_W    + NASSIGN)   // 98433

__device__ inline v16bf pack16(const float4& a, const float4& b,
                               const float4& c, const float4& d) {
  v16bf r;
  r[0]=(__bf16)a.x;  r[1]=(__bf16)a.y;  r[2]=(__bf16)a.z;  r[3]=(__bf16)a.w;
  r[4]=(__bf16)b.x;  r[5]=(__bf16)b.y;  r[6]=(__bf16)b.z;  r[7]=(__bf16)b.w;
  r[8]=(__bf16)c.x;  r[9]=(__bf16)c.y;  r[10]=(__bf16)c.z; r[11]=(__bf16)c.w;
  r[12]=(__bf16)d.x; r[13]=(__bf16)d.y; r[14]=(__bf16)d.z; r[15]=(__bf16)d.w;
  return r;
}

// One-time: wg fp32 [128,4096] -> bf16 in ws (1 MB). Keeps B fragment loads
// as raw 32B bf16 vectors (no per-step cvt, half the load count).
__global__ void wg_to_bf16(const float* __restrict__ wg, __bf16* __restrict__ wgbf) {
  const size_t i = (size_t)blockIdx.x * blockDim.x + threadIdx.x; // per float4
  const float4 v = ((const float4*)wg)[i];
  __bf16* o = wgbf + i * 4;
  o[0] = (__bf16)v.x; o[1] = (__bf16)v.y; o[2] = (__bf16)v.z; o[3] = (__bf16)v.w;
}

__global__ void gate_zero(int* __restrict__ counts) {
  if (threadIdx.x < NEXP) counts[threadIdx.x] = 0;
}

// 256 blocks x 256 threads (8 waves). Block computes 64 tokens x 128 experts.
// Wave w: M-tile = w>>1 (16 rows), N-tiles nbase..nbase+3 (nbase = (w&1)*4).
// K-loop is software-pipelined (unroll 2, double-buffered registers) so loads
// for step k+1 are in flight while step k converts/wmma's.
__global__ __launch_bounds__(256) void gate_gemm_route(
    const float* __restrict__ x, const __bf16* __restrict__ wgbf,
    float* __restrict__ out, int* __restrict__ experts,
    int* __restrict__ counts, float* __restrict__ me_partial)
{
  __shared__ float lg[TOK_PER_BLK][NEXP];   // 32 KB logits tile
  __shared__ float mx[TOK_PER_BLK];
  __shared__ float se[TOK_PER_BLK];
  __shared__ int   lcnt[NEXP];

  const int tid   = threadIdx.x;
  const int lane  = tid & 31;
  const int wave  = tid >> 5;
  const int mtile = wave >> 1;
  const int nbase = (wave & 1) * 4;
  const int blk   = blockIdx.x;

  if (tid < NEXP) lcnt[tid] = 0;

  // 16-bit A 16x32 layout (matches B): lanes0-15 -> K 0..15; lanes16-31 -> K 16..31
  const int rowL   = mtile * 16 + (lane & 15);
  const int grow   = blk * TOK_PER_BLK + rowL;
  const int kbaseA = (lane < 16) ? 0 : 16;
  // 16-bit B 32x16 layout (contiguous k, matches A): lanes0-15 -> K 0..15; lanes16-31 -> K 16..31
  const int kbaseB = (lane < 16) ? 0 : 16;
  const float* xrow = x + (size_t)grow * DMODEL;

  int coln[4];
#pragma unroll
  for (int n = 0; n < 4; ++n) coln[n] = (nbase + n) * 16 + (lane & 15);

  v8f acc[4] = {};   // 4 N-tiles x 8 VGPR f32 accumulator

  float4 A0[4], A1[4];
  v16bf  B0[4], B1[4];

  auto loadA = [&](int kk, float4* A) {
    const float4* g1 = (const float4*)(xrow + kk + kbaseA);
    const float4* g2 = (const float4*)(xrow + kk + 8 + kbaseA);
    A[0] = g1[0]; A[1] = g1[1]; A[2] = g2[0]; A[3] = g2[1];
  };
  auto loadB = [&](int kk, v16bf* B) {
#pragma unroll
    for (int n = 0; n < 4; ++n)
      B[n] = *(const v16bf*)(wgbf + (size_t)coln[n] * DMODEL + kk + kbaseB);
  };
  auto compute = [&](const float4* A, const v16bf* B) {
    v16bf af = pack16(A[0], A[1], A[2], A[3]);
#pragma unroll
    for (int n = 0; n < 4; ++n)
      acc[n] = __builtin_amdgcn_wmma_f32_16x16x32_bf16(
          false, af, false, B[n], (short)0, acc[n], false, false);
  };

  loadA(0, A0); loadB(0, B0);
  for (int kk = 0; kk < DMODEL; kk += 64) {
    __builtin_prefetch(xrow + kk + 128, 0, 0);   // pull next x cachelines
    loadA(kk + 32, A1); loadB(kk + 32, B1);
    compute(A0, B0);
    if (kk + 64 < DMODEL) { loadA(kk + 64, A0); loadB(kk + 64, B0); }
    compute(A1, B1);
  }

  // D layout: VGPR v -> M = v + (lane<16?0:8), N = lane&15 (within tile)
  const int rofs = (lane < 16) ? 0 : 8;
#pragma unroll
  for (int n = 0; n < 4; ++n) {
    const int col = coln[n];
#pragma unroll
    for (int v = 0; v < 8; ++v)
      lg[mtile * 16 + rofs + v][col] = acc[n][v];
  }
  __syncthreads();

  // Per-token: top-2 (raw logits, tie -> lower index) + softmax max/denominator
  if (tid < TOK_PER_BLK) {
    const int t = tid;
    float best = -INFINITY, sec = -INFINITY;
    int   bi = 0, si = 0;
    for (int e = 0; e < NEXP; ++e) {
      float v = lg[t][e];
      if (v > best)      { sec = best; si = bi; best = v; bi = e; }
      else if (v > sec)  { sec = v; si = e; }
    }
    mx[t] = best;
    float s = 0.f;
    for (int e = 0; e < NEXP; ++e) s += expf(lg[t][e] - best);
    se[t] = s;
    const int gt = blk * TOK_PER_BLK + t;
    experts[gt * 2 + 0] = bi;
    experts[gt * 2 + 1] = si;
    out[O_W + gt * 2 + 0] = best;
    out[O_W + gt * 2 + 1] = sec;
    atomicAdd(&lcnt[bi], 1);
    atomicAdd(&lcnt[si], 1);
  }
  __syncthreads();

  // Deterministic per-block softmax-mean partials (fixed summation order)
  if (tid < NEXP) {
    float s = 0.f;
    for (int t = 0; t < TOK_PER_BLK; ++t) s += expf(lg[t][tid] - mx[t]) / se[t];
    me_partial[blk * NEXP + tid] = s;
    atomicAdd(&counts[tid], lcnt[tid]);   // integer atomics: deterministic
  }
}

// 1 block x 128 threads: reduce me partials, cumsums, l_aux.
__global__ __launch_bounds__(128) void gate_scan(
    float* __restrict__ out, const int* __restrict__ counts,
    int* __restrict__ offsets, const float* __restrict__ me_partial)
{
  __shared__ float me_s[NEXP];
  __shared__ int   cnt_s[NEXP];
  const int e = threadIdx.x;
  float s = 0.f;
  for (int b = 0; b < NBLK1; ++b) s += me_partial[b * NEXP + e];
  me_s[e]  = s / (float)T_TOK;
  cnt_s[e] = counts[e];
  __syncthreads();
  if (e == 0) {
    int off = 0, bsum = 0;
    float laux = 0.f;
    for (int i = 0; i < NEXP; ++i) {
      const int c = cnt_s[i];
      offsets[i] = off; off += c;
      const int tpe = c < CAP ? c : CAP;
      bsum += tpe;
      out[O_BINS + i] = (float)bsum;
      out[O_TPE  + i] = (float)tpe;
      laux += me_s[i] * ((float)c / (float)T_TOK);
    }
    // mean(me*ce)*E*E/K = sum(me*ce)*E/K
    out[O_LAUX] = laux * ((float)NEXP / (float)TOPK);
  }
}

// One block per expert: gather assignments, O(n^2) stable rank
// (weight desc, flat index asc), scatter order/bin_ids.
__global__ __launch_bounds__(256) void gate_sort(
    const int* __restrict__ experts, float* __restrict__ out,
    const int* __restrict__ offsets)
{
  __shared__ float lw[MAXPER];
  __shared__ int   lid[MAXPER];
  __shared__ int   nsh;
  const int e = blockIdx.x;
  if (threadIdx.x == 0) nsh = 0;
  __syncthreads();
  for (int i = threadIdx.x; i < NASSIGN; i += 256) {
    if (experts[i] == e) {
      int p = atomicAdd(&nsh, 1);
      if (p < MAXPER) { lw[p] = out[O_W + i]; lid[p] = i; }
    }
  }
  __syncthreads();
  const int m    = nsh < MAXPER ? nsh : MAXPER;
  const int base = offsets[e];
  for (int i = threadIdx.x; i < m; i += 256) {
    const float wi = lw[i];
    const int  idi = lid[i];
    int r = 0;
    for (int j = 0; j < m; ++j) {
      const float wj = lw[j];
      r += (wj > wi) || (wj == wi && lid[j] < idi);
    }
    out[O_IDX  + base + r] = (float)idi;
    out[O_BIDS + base + r] = (float)e;
  }
}

extern "C" void kernel_launch(void* const* d_in, const int* in_sizes, int n_in,
                              void* d_out, int out_size, void* d_ws, size_t ws_size,
                              hipStream_t stream) {
  const float* x  = (const float*)d_in[0];   // [16384, 4096] fp32
  const float* wg = (const float*)d_in[1];   // [128, 4096] fp32
  float* out = (float*)d_out;

  char* ws = (char*)d_ws;
  size_t o = 0;
  int*   experts    = (int*)(ws + o);  o += (size_t)NASSIGN * 4;        // 128 KB
  int*   counts     = (int*)(ws + o);  o += (size_t)NEXP * 4;
  int*   offsets    = (int*)(ws + o);  o += (size_t)NEXP * 4;
  float* me_partial = (float*)(ws + o); o += (size_t)NBLK1 * NEXP * 4;  // 128 KB
  __bf16* wgbf      = (__bf16*)(ws + o);                                // 1 MB

  wg_to_bf16<<<(NEXP * DMODEL / 4) / 256, 256, 0, stream>>>(wg, wgbf);
  gate_zero<<<1, 128, 0, stream>>>(counts);
  gate_gemm_route<<<NBLK1, 256, 0, stream>>>(x, wgbf, out, experts, counts, me_partial);
  gate_scan<<<1, 128, 0, stream>>>(out, counts, offsets, me_partial);
  gate_sort<<<NEXP, 256, 0, stream>>>(experts, out, offsets);
}
